// CumulativeFFT_50294067036689
// MI455X (gfx1250) — matrix operand order repaired and checked by reading, not compiled, not run on hardware
//
#include <hip/hip_runtime.h>
#include <hip/hip_bf16.h>

// CumulativeFFT on MI455X (gfx1250), WMMA bf16 formulation, 32-row chunks.
//
// out[b,t,d,k,c] = pos_norm[t] * sum_{s<=t} x[b,s,d] * tw[s,k,c]
//
// Store-bound: 256 MB bf16 output (~11 us floor @ 23.3 TB/s). Compute is a
// blocked triangular GEMM on v_wmma_f32_16x16x32_bf16: per 32-t chunk and
// 16-wide kc tile, two WMMAs sharing one full-K=32 B operand, with the running
// cumsum carried through the WMMA C operand (50% MAC utilization vs 27% for
// 16-row tiles, and half the A/B operand-build traffic per output).

#define T_SEQ   1024
#define D_MODEL 512
#define BATCH   4
#define KC      64   // K_MAX * 2 (cos/sin interleaved, matches memory order)

typedef __attribute__((ext_vector_type(16))) __bf16 v16bf;
typedef __attribute__((ext_vector_type(8)))  float  v8f;

union V16 {
    v16bf        v;
    unsigned int w[8];
};

__device__ __forceinline__ unsigned short f32_to_bf16_rne(float f) {
    unsigned int u = __builtin_bit_cast(unsigned int, f);
    u += 0x7FFFu + ((u >> 16) & 1u);
    return (unsigned short)(u >> 16);
}

__device__ __forceinline__ float bf16_to_f32(unsigned short h) {
    unsigned int u = ((unsigned int)h) << 16;
    return __builtin_bit_cast(float, u);
}

// One wave == one (b, d) pair. 4 waves / block.
__global__ __launch_bounds__(128) void cumfft_wmma_kernel(
    const unsigned short* __restrict__ x,    // [B][T][D]     bf16 raw
    const unsigned short* __restrict__ tw,   // [T][KC]       bf16 raw
    const unsigned short* __restrict__ pn,   // [T]           bf16 raw
    unsigned short*       __restrict__ out)  // [B][T][D][KC] bf16 raw
{
    const int lane = threadIdx.x & 31;
    const int wave = threadIdx.x >> 5;
    const int col  = lane & 15;   // N column (B/C/D) and row index m (A)
    const int half = lane >> 4;   // lane group 0: lanes 0-15, 1: lanes 16-31

    const int d = blockIdx.x * 4 + wave;   // 0..511
    const int b = blockIdx.y;              // 0..3

    const unsigned short* xb = x + (size_t)b * T_SEQ * D_MODEL + d;

    float carry[4] = {0.f, 0.f, 0.f, 0.f};  // running cumsum per kc-tile (col = lane%16)

    for (int t0 = 0; t0 < T_SEQ; t0 += 32) {
        // ---- A operands for a 32-row chunk (s-offsets k = 0..31) ----
        // ISA A layout (16x32 bf16): lane m holds K {0..7}(V0-3) + {16..23}(V4-7);
        // lane m+16 holds K {8..15} + {24..31}; pairs packed lo/hi in each VGPR.
        //
        //   A_lo: rows t0+m,    A_lo[m,k] = x[t0+k]      if k <= m      (k 16..31 -> 0)
        //   A_hi: rows t0+16+m, A_hi[m,k] = x[t0+k]      if k <= m+16
        //         => k 0..15 always included; k = 16+j masked j <= m.
        V16 Alo, Ahi;
#pragma unroll
        for (int v = 0; v < 4; ++v) {
            const int j0 = 8 * half + 2 * v;           // k slot within each 16-group
            unsigned int xa0 = xb[(size_t)(t0 + j0)          * D_MODEL];
            unsigned int xa1 = xb[(size_t)(t0 + j0 + 1)      * D_MODEL];
            unsigned int xb0 = xb[(size_t)(t0 + 16 + j0)     * D_MODEL];
            unsigned int xb1 = xb[(size_t)(t0 + 16 + j0 + 1) * D_MODEL];
            const unsigned int pk_a = xa0 | (xa1 << 16);          // unmasked k 0..15 pair
            unsigned int m_a0 = (j0     <= col) ? xa0 : 0u;       // triangular masks
            unsigned int m_a1 = (j0 + 1 <= col) ? xa1 : 0u;
            unsigned int m_b0 = (j0     <= col) ? xb0 : 0u;
            unsigned int m_b1 = (j0 + 1 <= col) ? xb1 : 0u;
            Alo.w[v]     = m_a0 | (m_a1 << 16);   // rows 0..15: tri over k 0..15
            Alo.w[v + 4] = 0u;                    // rows 0..15: k 16..31 never used
            Ahi.w[v]     = pk_a;                  // rows 16..31: k 0..15 full
            Ahi.w[v + 4] = m_b0 | (m_b1 << 16);   // rows 16..31: tri over k 16..31
        }

        // ---- pos_norm for the 16 rows this lane's D registers cover ----
        float pnv[16];
#pragma unroll
        for (int r = 0; r < 8; ++r) {
            pnv[r]     = bf16_to_f32(pn[t0 + r + 8 * half]);
            pnv[r + 8] = bf16_to_f32(pn[t0 + 16 + r + 8 * half]);
        }

        // Prefetch next chunk of x (uniform branch; emits global_prefetch_b8).
        if (t0 + 32 < T_SEQ) {
            __builtin_prefetch(&xb[(size_t)(t0 + 32 + 2 * lane) * D_MODEL], 0, 1);
        }

#pragma unroll
        for (int kt = 0; kt < 4; ++kt) {
            const int kc0 = kt * 16;

            // ---- B (32x16 bf16): B[k,n] = tw[t0+k, kc0+n], full K = 32 ----
            // Layout (matches A): lanes 0-15 hold K rows 0..7,16..23,
            // lanes 16-31 hold K rows 8..15,24..31; 2 rows per VGPR (lo/hi).
            V16 Bb;
#pragma unroll
            for (int v = 0; v < 8; ++v) {
                const int s = t0 + 8 * half + 2 * v + 8 * (v >> 2);
                unsigned int b0 = tw[(size_t)s       * KC + kc0 + col];
                unsigned int b1 = tw[(size_t)(s + 1) * KC + kc0 + col];
                Bb.w[v] = b0 | (b1 << 16);
            }

            // C = carry broadcast into every row.
            v8f c;
#pragma unroll
            for (int r = 0; r < 8; ++r) c[r] = carry[kt];

            // Rows t0..t0+15 (triangular prefix) and t0+16..t0+31 (full + tri).
            v8f acc_lo = __builtin_amdgcn_wmma_f32_16x16x32_bf16(
                false, Alo.v, false, Bb.v, (short)0, c, false, false);
            v8f acc_hi = __builtin_amdgcn_wmma_f32_16x16x32_bf16(
                false, Ahi.v, false, Bb.v, (short)0, c, false, false);

            // New carry = chunk row 31 = acc_hi[7] in lanes 16..31.
            carry[kt] = __shfl(acc_hi[7], 16 + col, 32);

            // Scale by pos_norm[t] and store bf16.
#pragma unroll
            for (int r = 0; r < 8; ++r) {
                const int t_lo = t0 + r + 8 * half;        // VGPR r <-> rows r / r+8
                const int t_hi = t_lo + 16;
                out[(((size_t)b * T_SEQ + t_lo) * D_MODEL + d) * KC + kc0 + col] =
                    f32_to_bf16_rne(acc_lo[r] * pnv[r]);
                out[(((size_t)b * T_SEQ + t_hi) * D_MODEL + d) * KC + kc0 + col] =
                    f32_to_bf16_rne(acc_hi[r] * pnv[r + 8]);
            }
        }
    }
}

extern "C" void kernel_launch(void* const* d_in, const int* in_sizes, int n_in,
                              void* d_out, int out_size, void* d_ws, size_t ws_size,
                              hipStream_t stream) {
    (void)in_sizes; (void)n_in; (void)out_size; (void)d_ws; (void)ws_size;
    const unsigned short* x  = (const unsigned short*)d_in[0];  // bf16 raw
    const unsigned short* tw = (const unsigned short*)d_in[1];  // bf16 raw
    const unsigned short* pn = (const unsigned short*)d_in[2];  // bf16 raw
    unsigned short* out = (unsigned short*)d_out;               // bf16 raw

    dim3 grid(D_MODEL / 4, BATCH);   // (128, 4) blocks, 4 waves each -> 2048 waves
    cumfft_wmma_kernel<<<grid, 128, 0, stream>>>(x, tw, pn, out);
}
